// TRMNeuralSymbolicSolver_35820027249038
// MI455X (gfx1250) — matrix operand order repaired and checked by reading, not compiled, not run on hardware
//
#include <hip/hip_runtime.h>
#include <hip/hip_bf16.h>

typedef __attribute__((ext_vector_type(16))) _Float16 v16h;
typedef __attribute__((ext_vector_type(8)))  _Float16 v8h;
typedef __attribute__((ext_vector_type(8)))  float    v8f;

#define N_BATCH   8192
#define N_CELLS   900     // 30*30
#define N_COLORS  10
#define D_HID     128
#define D_A       128
#define D_Z       64
#define D_GF      16
#define D_RH      64
#define D_NT      64
#define D_CAT     224     // 208 (=128+64+16) zero-padded to a multiple of 32

__device__ __forceinline__ float gelu_exact(float x) {
  // torch nn.GELU default (exact erf), matches jax.nn.gelu(approximate=False)
  return 0.5f * x * (1.0f + erff(x * 0.70710678118654752440f));
}

// ---- repack f32 weight [K,N] -> f16 transposed, K zero-padded: [N, Kpad] ----
__global__ void k_prep_weight(const float* __restrict__ w, _Float16* __restrict__ wt,
                              int K, int N, int Kpad) {
  int i = blockIdx.x * blockDim.x + threadIdx.x;
  if (i >= N * Kpad) return;
  int n = i / Kpad, k = i - n * Kpad;
  wt[i] = (k < K) ? (_Float16)w[(size_t)k * N + n] : (_Float16)0.0f;
}

__global__ void k_f32_to_f16(const float* __restrict__ src, _Float16* __restrict__ dst, int n) {
  int i = blockIdx.x * blockDim.x + threadIdx.x;
  if (i < n) dst[i] = (_Float16)src[i];
}

// ---- encoder layer 1: one-hot @ enc_w1 == gather-sum of enc_w1 rows --------
// one wave per batch row; lane owns 4 of 128 features (coalesced 512B/row reads)
__global__ void k_encoder_gather(const int* __restrict__ grid,
                                 const float* __restrict__ w1,
                                 const float* __restrict__ b1,
                                 _Float16* __restrict__ h) {
  int wave = (int)((blockIdx.x * blockDim.x + threadIdx.x) >> 5);
  int lane = threadIdx.x & 31;
  if (wave >= N_BATCH) return;
  const int* g = grid + (size_t)wave * N_CELLS;
  int f = lane * 4;
  float a0 = 0.f, a1 = 0.f, a2 = 0.f, a3 = 0.f;
  for (int cell = 0; cell < N_CELLS; ++cell) {
    if ((cell & 63) == 0) __builtin_prefetch(g + cell + 64, 0, 0);  // global_prefetch_b8
    int color = g[cell];
    const float4 w = *(const float4*)(w1 + (size_t)(cell * N_COLORS + color) * D_HID + f);
    a0 += w.x; a1 += w.y; a2 += w.z; a3 += w.w;
  }
  _Float16* hp = h + (size_t)wave * D_HID + f;
  hp[0] = (_Float16)gelu_exact(a0 + b1[f + 0]);
  hp[1] = (_Float16)gelu_exact(a1 + b1[f + 1]);
  hp[2] = (_Float16)gelu_exact(a2 + b1[f + 2]);
  hp[3] = (_Float16)gelu_exact(a3 + b1[f + 3]);
}

// ---- generic tall GEMM: out = act(A[rows,Kpad] @ Wt^T + bias) via WMMA -----
// Wt is pre-transposed f16 [N, Kpad]. One wave per 16x16 output tile.
// Tile guard is wave-uniform so EXEC is all-1s at every v_wmma (ISA req).
__global__ void k_gemm_wmma(const _Float16* __restrict__ A,
                            const _Float16* __restrict__ Wt,
                            const float*    __restrict__ bias,
                            _Float16* __restrict__ out16,   // may be null
                            float*    __restrict__ out32,   // may be null
                            int rows, int N, int Kpad, int do_gelu) {
  int wave = (int)((blockIdx.x * blockDim.x + threadIdx.x) >> 5);
  int lane = threadIdx.x & 31;
  int ntile = N >> 4;
  int tiles = (rows >> 4) * ntile;
  if (wave >= tiles) return;                 // wave-uniform
  int rt = wave / ntile;
  int ct = wave - rt * ntile;
  int row0 = rt << 4, col0 = ct << 4;
  int lo = lane & 15, hi = lane >> 4;

  // A fragment (contiguous k order, same as B): lane<16 -> K{k0..k0+15},
  //                               lane>=16 -> K{k0+16..k0+31}
  const _Float16* ap = A  + (size_t)(row0 + lo) * Kpad + hi * 16;
  // B fragment: lane n holds column n; lanes 0-15 K{k0..k0+15}, 16-31 K{k0+16..k0+31}
  const _Float16* bp = Wt + (size_t)(col0 + lo) * Kpad + hi * 16;

  v8f acc = {};
  for (int k0 = 0; k0 < Kpad; k0 += 32) {
    v8h x0 = *(const v8h*)(ap + k0);
    v8h x1 = *(const v8h*)(ap + k0 + 8);
    v16h a = __builtin_shufflevector(x0, x1, 0,1,2,3,4,5,6,7,8,9,10,11,12,13,14,15);
    v16h b = *(const v16h*)(bp + k0);
    // (neg_a, A, neg_b, B, c_mod, C, reuse_a, reuse_b)
    acc = __builtin_amdgcn_wmma_f32_16x16x32_f16(false, a, false, b,
                                                 (short)0, acc, false, false);
  }

  // C layout: VGPR r -> M = (lane/16)*8 + r, N = lane%16
  int n  = col0 + lo;
  int mb = row0 + hi * 8;
  float bv = bias[n];
#pragma unroll
  for (int r = 0; r < 8; ++r) {
    float v = acc[r] + bv;
    if (do_gelu) v = gelu_exact(v);
    if (out16) out16[(size_t)(mb + r) * N + n] = (_Float16)v;
    if (out32) out32[(size_t)(mb + r) * N + n] = v;
  }
}

// ---- concat builder for refiner inputs (zero-pads 208 -> 224) --------------
// mode 0: [y(128), z(64), gf(16), 0*16] ; mode 1: [z(64), y(128), gf(16), 0*16]
__global__ void k_concat(const _Float16* __restrict__ y, const _Float16* __restrict__ z,
                         const _Float16* __restrict__ gf, _Float16* __restrict__ cat,
                         int mode) {
  int i = blockIdx.x * blockDim.x + threadIdx.x;
  if (i >= N_BATCH * D_CAT) return;
  int b = i / D_CAT, j = i - b * D_CAT;
  _Float16 v = (_Float16)0.0f;
  if (mode == 0) {
    if (j < 128)      v = y[(size_t)b * 128 + j];
    else if (j < 192) v = z[(size_t)b * 64 + (j - 128)];
    else if (j < 208) v = gf[(size_t)b * 16 + (j - 192)];
  } else {
    if (j < 64)       v = z[(size_t)b * 64 + j];
    else if (j < 192) v = y[(size_t)b * 128 + (j - 64)];
    else if (j < 208) v = gf[(size_t)b * 16 + (j - 192)];
  }
  cat[i] = v;
}

// ---- fused gumbel-softmax over 64 templates (tau = 1.0), one wave per row --
__global__ void k_softmax64(const float* __restrict__ logits, const float* __restrict__ gum,
                            float* __restrict__ out) {
  int wave = (int)((blockIdx.x * blockDim.x + threadIdx.x) >> 5);
  int lane = threadIdx.x & 31;
  if (wave >= N_BATCH) return;
  size_t base = (size_t)wave * 64;
  float v0 = logits[base + lane]      + gum[base + lane];
  float v1 = logits[base + lane + 32] + gum[base + lane + 32];
  float m = fmaxf(v0, v1);
  for (int off = 16; off; off >>= 1) m = fmaxf(m, __shfl_xor(m, off, 32));
  float e0 = expf(v0 - m), e1 = expf(v1 - m);
  float s = e0 + e1;
  for (int off = 16; off; off >>= 1) s += __shfl_xor(s, off, 32);
  float inv = 1.0f / s;
  out[base + lane]      = e0 * inv;
  out[base + lane + 32] = e1 * inv;
}

// ---------------------------------------------------------------------------
extern "C" void kernel_launch(void* const* d_in, const int* in_sizes, int n_in,
                              void* d_out, int out_size, void* d_ws, size_t ws_size,
                              hipStream_t stream) {
  (void)in_sizes; (void)n_in; (void)out_size; (void)ws_size;
  const int*   grid    = (const int*)  d_in[0];
  const float* gfeat   = (const float*)d_in[1];
  const float* gumbel  = (const float*)d_in[2];
  const float* enc_w1  = (const float*)d_in[3];
  const float* enc_b1  = (const float*)d_in[4];
  const float* enc_w2  = (const float*)d_in[5];
  const float* enc_b2  = (const float*)d_in[6];
  const float* wy      = (const float*)d_in[7];
  const float* by      = (const float*)d_in[8];
  const float* wz      = (const float*)d_in[9];
  const float* bz      = (const float*)d_in[10];
  const float* rz_w1   = (const float*)d_in[11];
  const float* rz_b1   = (const float*)d_in[12];
  const float* rz_w2   = (const float*)d_in[13];
  const float* rz_b2   = (const float*)d_in[14];
  const float* ry_w1   = (const float*)d_in[15];
  const float* ry_b1   = (const float*)d_in[16];
  const float* ry_w2   = (const float*)d_in[17];
  const float* ry_b2   = (const float*)d_in[18];
  const float* sel_w1  = (const float*)d_in[19];
  const float* sel_b1  = (const float*)d_in[20];
  const float* sel_w2  = (const float*)d_in[21];
  const float* sel_b2  = (const float*)d_in[22];
  float* out = (float*)d_out;

  // ---- workspace carve-up (256B aligned slabs) ----
  char* p = (char*)d_ws;
  auto alloc = [&](size_t bytes) -> char* {
    char* r = p; p += (bytes + 255) & ~(size_t)255; return r;
  };
  _Float16* wt_enc2 = (_Float16*)alloc((size_t)D_HID * D_HID * 2);
  _Float16* wt_wy   = (_Float16*)alloc((size_t)D_A   * D_HID * 2);
  _Float16* wt_wz   = (_Float16*)alloc((size_t)D_Z   * D_HID * 2);
  _Float16* wt_rz1  = (_Float16*)alloc((size_t)D_RH  * D_CAT * 2);
  _Float16* wt_rz2  = (_Float16*)alloc((size_t)D_Z   * D_RH  * 2);
  _Float16* wt_ry1  = (_Float16*)alloc((size_t)D_RH  * D_CAT * 2);
  _Float16* wt_ry2  = (_Float16*)alloc((size_t)D_A   * D_RH  * 2);
  _Float16* wt_sel1 = (_Float16*)alloc((size_t)D_RH  * D_A   * 2);
  _Float16* wt_sel2 = (_Float16*)alloc((size_t)D_NT  * D_RH  * 2);
  _Float16* gf16    = (_Float16*)alloc((size_t)N_BATCH * D_GF  * 2);
  _Float16* h0      = (_Float16*)alloc((size_t)N_BATCH * D_HID * 2);
  _Float16* h1      = (_Float16*)alloc((size_t)N_BATCH * D_HID * 2);
  _Float16* ybuf    = (_Float16*)alloc((size_t)N_BATCH * D_A   * 2);
  _Float16* zbuf    = (_Float16*)alloc((size_t)N_BATCH * D_Z   * 2);
  _Float16* cat     = (_Float16*)alloc((size_t)N_BATCH * D_CAT * 2);
  _Float16* rh      = (_Float16*)alloc((size_t)N_BATCH * D_RH  * 2);
  float*    logits  = (float*)   alloc((size_t)N_BATCH * D_NT  * 4);

  const int TPB = 256;
  auto blocks1d = [](int n, int tpb) { return (n + tpb - 1) / tpb; };

  // ---- stage 0: repack weights to f16 [N, Kpad] ----
  struct { const float* w; _Float16* wt; int K, N, Kpad; } prep[9] = {
    { enc_w2, wt_enc2, D_HID, D_HID, D_HID },
    { wy,     wt_wy,   D_HID, D_A,   D_HID },
    { wz,     wt_wz,   D_HID, D_Z,   D_HID },
    { rz_w1,  wt_rz1,  208,   D_RH,  D_CAT },
    { rz_w2,  wt_rz2,  D_RH,  D_Z,   D_RH  },
    { ry_w1,  wt_ry1,  208,   D_RH,  D_CAT },
    { ry_w2,  wt_ry2,  D_RH,  D_A,   D_RH  },
    { sel_w1, wt_sel1, D_A,   D_RH,  D_A   },
    { sel_w2, wt_sel2, D_RH,  D_NT,  D_RH  },
  };
  for (int i = 0; i < 9; ++i) {
    int n = prep[i].N * prep[i].Kpad;
    k_prep_weight<<<blocks1d(n, TPB), TPB, 0, stream>>>(prep[i].w, prep[i].wt,
                                                        prep[i].K, prep[i].N, prep[i].Kpad);
  }
  k_f32_to_f16<<<blocks1d(N_BATCH * D_GF, TPB), TPB, 0, stream>>>(gfeat, gf16, N_BATCH * D_GF);

  // ---- stage 1: encoder layer 1 (one-hot matmul as gather-sum) + GELU ----
  k_encoder_gather<<<N_BATCH / 8, TPB, 0, stream>>>(grid, enc_w1, enc_b1, h0);

  // helper for gemm launches: one wave (32 thr) per 16x16 tile, 8 waves/block
  auto gemm = [&](const _Float16* A, const _Float16* Wt, const float* bias,
                  _Float16* o16, float* o32, int N, int Kpad, int act) {
    int tiles = (N_BATCH / 16) * (N / 16);
    k_gemm_wmma<<<(tiles + 7) / 8, TPB, 0, stream>>>(A, Wt, bias, o16, o32,
                                                     N_BATCH, N, Kpad, act);
  };

  // ---- stage 2: encoder layer 2 + heads ----
  gemm(h0, wt_enc2, enc_b2, h1,   nullptr, D_HID, D_HID, 1);  // h = gelu(h@w2+b2)
  gemm(h1, wt_wy,   by,     ybuf, nullptr, D_A,   D_HID, 0);  // y
  gemm(h1, wt_wz,   bz,     zbuf, nullptr, D_Z,   D_HID, 0);  // z

  // ---- stage 3: T=3 refinement cycles (stop_gradient is a no-op in fwd) ----
  const int CATN = N_BATCH * D_CAT;
  for (int t = 0; t < 3; ++t) {
    k_concat<<<blocks1d(CATN, TPB), TPB, 0, stream>>>(ybuf, zbuf, gf16, cat, 0);
    gemm(cat, wt_rz1, rz_b1, rh,   nullptr, D_RH, D_CAT, 1);
    gemm(rh,  wt_rz2, rz_b2, zbuf, nullptr, D_Z,  D_RH,  0);
    k_concat<<<blocks1d(CATN, TPB), TPB, 0, stream>>>(ybuf, zbuf, gf16, cat, 1);
    gemm(cat, wt_ry1, ry_b1, rh,   nullptr, D_RH, D_CAT, 1);
    gemm(rh,  wt_ry2, ry_b2, ybuf, nullptr, D_A,  D_RH,  0);
  }

  // ---- stage 4: selector + gumbel softmax ----
  gemm(ybuf, wt_sel1, sel_b1, rh, nullptr, D_RH, D_A,  1);
  gemm(rh,   wt_sel2, sel_b2, nullptr, logits, D_NT, D_RH, 0);
  k_softmax64<<<N_BATCH / 8, TPB, 0, stream>>>(logits, gumbel, out);
}
